// ResidualQuantizer_26740466385191
// MI455X (gfx1250) — matrix operand order repaired and checked by reading, not compiled, not run on hardware
//
#include <hip/hip_runtime.h>

// ---------------------------------------------------------------------------
// Residual VQ (3 levels: K=4,16,256; D=32) for MI455X (gfx1250, wave32).
// Scores via chained v_wmma_f32_16x16x32_bf16 in split (hi+lo) bf16 precision:
//   score = Alo*Bhi + Ahi*Blo + Ahi*Bhi   (f32 accumulate through WMMA C)
// giving ~2^-17 relative accuracy, so argmin matches fp32 reference except on
// near-exact ties. B fragments pre-split & cached in LDS (36KB). Argmax is a
// branchless u64 (ordered-score | ~index) max. Memory-bound: ~270MB @23.3TB/s.
// ---------------------------------------------------------------------------

typedef __attribute__((ext_vector_type(16))) __bf16   v16bf;
typedef __attribute__((ext_vector_type(8)))  float    v8f;
typedef __attribute__((ext_vector_type(8)))  unsigned v8u;

union BF16Frag { v16bf h; v8u u; };

__device__ __forceinline__ unsigned f2bf(float f) {
    unsigned u = __float_as_uint(f);
    u += 0x7FFFu + ((u >> 16) & 1u);          // round-to-nearest-even
    return u >> 16;
}
__device__ __forceinline__ unsigned pack2(float lo, float hi) {
    return f2bf(lo) | (f2bf(hi) << 16);
}
__device__ __forceinline__ unsigned long long shflxor_u64(unsigned long long v, int m) {
    int lo = __shfl_xor((int)(v & 0xFFFFFFFFull), m, 32);
    int hi = __shfl_xor((int)(v >> 32), m, 32);
    return ((unsigned long long)(unsigned)hi << 32) | (unsigned)lo;
}

#define RVQ_BETA 0.25f

__launch_bounds__(256)
__global__ void ResidualQuantizer_26740466385191_kernel(
        const float* __restrict__ z,
        const float* __restrict__ cb0,
        const float* __restrict__ cb1,
        const float* __restrict__ cb2,
        float* __restrict__ out, int B)
{
    // 18 chunks x 32 lanes x (8 hi + 8 lo) packed-bf16 dwords
    __shared__ unsigned s_bfrag[18 * 32 * 16];       // 36 KB
    __shared__ float    s_bias[16 + 16 + 256];       // 0.5*||c||^2 (+inf pads)
    __shared__ int      s_code[8 * 16];

    const int tid = threadIdx.x;

    // ---- precompute 0.5*||c||^2 biases ------------------------------------
    for (int i = tid; i < 288; i += 256) {
        const float* cb; int k; bool pad = false;
        if (i < 16)      { cb = cb0; k = i;      pad = (i >= 4); }
        else if (i < 32) { cb = cb1; k = i - 16; }
        else             { cb = cb2; k = i - 32; }
        float v = 1e30f;
        if (!pad) {
            float s = 0.f;
            #pragma unroll
            for (int d = 0; d < 32; ++d) { float c = cb[k * 32 + d]; s += c * c; }
            v = 0.5f * s;
        }
        s_bias[i] = v;
    }

    // ---- precompute split-bf16 B fragments (CB^T columns) into LDS --------
    // chunk 0 -> level0 (K=4, clamped), chunk 1 -> level1, chunks 2..17 -> level2
    for (int i = tid; i < 576; i += 256) {
        const int chunk = i >> 5;
        const int ln    = i & 31;
        const int cc    = ln & 15;
        const int bd    = (ln < 16) ? 0 : 8;         // contraction-dim base
        const float* cb; int k;
        if (chunk == 0)      { cb = cb0; k = cc & 3; }
        else if (chunk == 1) { cb = cb1; k = cc; }
        else                 { cb = cb2; k = (chunk - 2) * 16 + cc; }
        unsigned* dst = &s_bfrag[i * 16];
        #pragma unroll
        for (int j = 0; j < 8; ++j) {
            float f0 = cb[k * 32 + bd + 2 * j + (j >= 4 ? 8 : 0)];
            float f1 = cb[k * 32 + bd + 2 * j + 1 + (j >= 4 ? 8 : 0)];
            unsigned h0 = f2bf(f0), h1 = f2bf(f1);
            dst[j]     = h0 | (h1 << 16);
            dst[8 + j] = pack2(f0 - __uint_as_float(h0 << 16),
                               f1 - __uint_as_float(h1 << 16));
        }
    }
    __syncthreads();

    const int wave    = tid >> 5;
    const int lane    = tid & 31;
    const int halfsel = lane >> 4;        // 0: dims 0-7/16-23, 1: dims 8-15/24-31
    const int col     = lane & 15;        // WMMA N column / row-in-tile
    const int c0      = halfsel * 8;      // A-fragment dim base
    const long tile   = (long)blockIdx.x * 8 + wave;
    const long grow   = tile * 16 + col;  // global row this lane owns

    // ---- load z into residual registers (A-fragment element order) --------
    float res[16], q[16];
    {
        const float4* p = (const float4*)(z + grow * 32);
        float4 a0 = p[c0 / 4 + 0];
        float4 a1 = p[c0 / 4 + 1];
        float4 a2 = p[c0 / 4 + 4];
        float4 a3 = p[c0 / 4 + 5];
        res[0]=a0.x;  res[1]=a0.y;  res[2]=a0.z;  res[3]=a0.w;
        res[4]=a1.x;  res[5]=a1.y;  res[6]=a1.z;  res[7]=a1.w;
        res[8]=a2.x;  res[9]=a2.y;  res[10]=a2.z; res[11]=a2.w;
        res[12]=a3.x; res[13]=a3.y; res[14]=a3.z; res[15]=a3.w;
        #pragma unroll
        for (int i = 0; i < 16; ++i) q[i] = 0.f;
    }

    float lsum = 0.f;
    const float* const cbs[3]  = { cb0, cb1, cb2 };
    const int          nch[3]  = { 1, 1, 16 };
    const int          boff[3] = { 0, 16, 32 };
    const int          cbase[3] = { 0, 1, 2 };

    #pragma unroll
    for (int level = 0; level < 3; ++level) {
        const float* cb = cbs[level];

        // A hi/lo fragments from the f32 residual
        BF16Frag Ahi, Alo;
        #pragma unroll
        for (int j = 0; j < 8; ++j) {
            float f0 = res[2 * j], f1 = res[2 * j + 1];
            unsigned h0 = f2bf(f0), h1 = f2bf(f1);
            Ahi.u[j] = h0 | (h1 << 16);
            Alo.u[j] = pack2(f0 - __uint_as_float(h0 << 16),
                             f1 - __uint_as_float(h1 << 16));
        }

        unsigned long long best[8];
        #pragma unroll
        for (int j = 0; j < 8; ++j) best[j] = 0ull;

        for (int ch = 0; ch < nch[level]; ++ch) {
            const int kcol = ch * 16 + col;            // candidate code (column)
            // B hi/lo fragments from LDS cache (4x ds_load_b128)
            const uint4* bp =
                (const uint4*)&s_bfrag[((cbase[level] + ch) * 32 + lane) * 16];
            uint4 h0 = bp[0], h1 = bp[1], l0 = bp[2], l1 = bp[3];
            BF16Frag Bhi, Blo;
            Bhi.u[0]=h0.x; Bhi.u[1]=h0.y; Bhi.u[2]=h0.z; Bhi.u[3]=h0.w;
            Bhi.u[4]=h1.x; Bhi.u[5]=h1.y; Bhi.u[6]=h1.z; Bhi.u[7]=h1.w;
            Blo.u[0]=l0.x; Blo.u[1]=l0.y; Blo.u[2]=l0.z; Blo.u[3]=l0.w;
            Blo.u[4]=l1.x; Blo.u[5]=l1.y; Blo.u[6]=l1.z; Blo.u[7]=l1.w;

            // split-precision product chained through the f32 accumulator
            v8f acc = {};
            acc = __builtin_amdgcn_wmma_f32_16x16x32_bf16(
                false, Alo.h, false, Bhi.h, (short)0, acc, false, false);
            acc = __builtin_amdgcn_wmma_f32_16x16x32_bf16(
                false, Ahi.h, false, Blo.h, (short)0, acc, false, false);
            acc = __builtin_amdgcn_wmma_f32_16x16x32_bf16(
                false, Ahi.h, false, Bhi.h, (short)0, acc, false, false);

            const float bias = s_bias[boff[level] + kcol];
            const unsigned ilow = ~(unsigned)kcol;     // ties -> lowest index
            #pragma unroll
            for (int j = 0; j < 8; ++j) {
                unsigned u  = __float_as_uint(acc[j] - bias);
                unsigned ou = u ^ ((unsigned)((int)u >> 31) | 0x80000000u);
                unsigned long long key = ((unsigned long long)ou << 32) | ilow;
                best[j] = (key > best[j]) ? key : best[j];   // branchless
            }
        }

        // branchless argmax across the 16 columns of each half-wave group
        #pragma unroll
        for (int m = 1; m < 16; m <<= 1) {
            #pragma unroll
            for (int j = 0; j < 8; ++j) {
                unsigned long long ok = shflxor_u64(best[j], m);
                best[j] = (ok > best[j]) ? ok : best[j];
            }
        }

        // publish per-row codes: slot j = row j (low group) / row j+8 (high)
        if (col == 0) {
            #pragma unroll
            for (int j = 0; j < 8; ++j)
                s_code[wave * 16 + halfsel * 8 + j] = (int)(~(unsigned)best[j]);
        }
        __syncthreads();
        const int code = s_code[wave * 16 + col];

        // codes output (float-cast), lanes 0-15 cover the 16 tile rows
        if (lane < 16) out[(long)level * B + grow] = (float)code;

        // gather winning codebook row, update residual / quantized / loss
        const float4* ep = (const float4*)(cb + (long)code * 32);
        float4 e0 = ep[c0 / 4 + 0];
        float4 e1 = ep[c0 / 4 + 1];
        float4 e2 = ep[c0 / 4 + 4];
        float4 e3 = ep[c0 / 4 + 5];
        float ev[16] = { e0.x,e0.y,e0.z,e0.w, e1.x,e1.y,e1.z,e1.w,
                         e2.x,e2.y,e2.z,e2.w, e3.x,e3.y,e3.z,e3.w };
        #pragma unroll
        for (int i = 0; i < 16; ++i) {
            q[i]   += ev[i];
            res[i] -= ev[i];
            lsum   += res[i] * res[i];   // (r - e)^2 with the new residual
        }
    }

    // quantized_total = e0+e1+e2
    {
        float4* qp = (float4*)(out + 3L * B + grow * 32);
        qp[c0 / 4 + 0] = make_float4(q[0],  q[1],  q[2],  q[3]);
        qp[c0 / 4 + 1] = make_float4(q[4],  q[5],  q[6],  q[7]);
        qp[c0 / 4 + 4] = make_float4(q[8],  q[9],  q[10], q[11]);
        qp[c0 / 4 + 5] = make_float4(q[12], q[13], q[14], q[15]);
    }

    // loss: (1+beta)/(B*D) * SSE, wave-reduced, one atomic per wave
    lsum *= (1.f + RVQ_BETA) / ((float)B * 32.f);
    #pragma unroll
    for (int m = 1; m < 32; m <<= 1) lsum += __shfl_xor(lsum, m, 32);
    if (lane == 0) atomicAdd(out + 35L * B, lsum);
}

extern "C" void kernel_launch(void* const* d_in, const int* in_sizes, int n_in,
                              void* d_out, int out_size, void* d_ws, size_t ws_size,
                              hipStream_t stream) {
    const float* z   = (const float*)d_in[0];
    const float* cb0 = (const float*)d_in[1];
    const float* cb1 = (const float*)d_in[2];
    const float* cb2 = (const float*)d_in[3];
    float* out = (float*)d_out;

    const int B = in_sizes[0] / 32;               // 1048576
    // zero the scalar-loss slot (harness poisons d_out; atomics accumulate)
    hipMemsetAsync(out + 35L * B, 0, sizeof(float), stream);

    const int rowsPerBlock = 8 * 16;              // 8 waves x 16-row tiles
    const int grid = B / rowsPerBlock;            // 8192
    ResidualQuantizer_26740466385191_kernel<<<grid, 256, 0, stream>>>(
        z, cb0, cb1, cb2, out, B);
}
